// AutoEncoderTopK_12249246728723
// MI455X (gfx1250) — matrix operand order repaired and checked by reading, not compiled, not run on hardware
//
#include <hip/hip_runtime.h>

// ---------------------------------------------------------------------------
// TopK sparse autoencoder forward for MI455X (gfx1250, wave32, WMMA).
// d_out = [reconstructed (16*512*1024 f32)] ++ [encoded (16*512*16384 f32)]
// ws layout: xb bf16 @0 (16MB) | W_enc bf16 @16MB (32MB) | W_dec^T f32 @48MB
//            (64MB) | topk idx @112MB (512KB)
// ---------------------------------------------------------------------------

#define AE_C   1024
#define AE_D   16384
#define AE_TOK 8192      // 16 * 512 tokens
#define AE_WIN 4
#define AE_NW  2048      // AE_TOK / AE_WIN windows
#define AE_K   64

typedef __attribute__((ext_vector_type(16))) __bf16 v16bf;
typedef __attribute__((ext_vector_type(8)))  float  v8f;

union BFrag { uint4 u[2]; v16bf v; };

__device__ __forceinline__ unsigned short f2bf(float f) {
  union { float f; unsigned u; } v; v.f = f;
  unsigned r = v.u + 0x7FFFu + ((v.u >> 16) & 1u);   // round-to-nearest-even
  return (unsigned short)(r >> 16);
}
__device__ __forceinline__ float bf2f(unsigned short h) {
  union { unsigned u; float f; } v; v.u = ((unsigned)h) << 16; return v.f;
}

// --------------------------- prep kernels ----------------------------------

__global__ __launch_bounds__(256) void ae_prep_x(const float* __restrict__ x,
                                                 const float* __restrict__ b_dec,
                                                 unsigned short* __restrict__ xb,
                                                 int n) {
  int i = blockIdx.x * 256 + threadIdx.x;
  if (i < n) xb[i] = f2bf(x[i] - b_dec[i & (AE_C - 1)]);
}

__global__ __launch_bounds__(256) void ae_prep_w(const float* __restrict__ w,
                                                 unsigned short* __restrict__ wb,
                                                 int n) {
  int i = blockIdx.x * 256 + threadIdx.x;
  if (i < n) wb[i] = f2bf(w[i]);
}

// W_dec is (C=1024, D=16384); produce W_dec^T (D, C) so the sparse decoder
// reads coalesced 1024-float rows per selected dictionary column.
__global__ __launch_bounds__(256) void ae_transpose_wdec(const float* __restrict__ wd,
                                                         float* __restrict__ wt) {
  __shared__ float tile[32][33];
  const int dx = blockIdx.x * 32;            // along D
  const int cy = blockIdx.y * 32;            // along C
  const int tx = threadIdx.x & 31;
  const int ty = threadIdx.x >> 5;           // 0..7
#pragma unroll
  for (int j = 0; j < 4; ++j)
    tile[ty + j * 8][tx] = wd[(size_t)(cy + ty + j * 8) * AE_D + dx + tx];
  __syncthreads();
#pragma unroll
  for (int j = 0; j < 4; ++j)
    wt[(size_t)(dx + ty + j * 8) * AE_C + cy + tx] = tile[tx][ty + j * 8];
}

// --------------------------- encoder (WMMA) --------------------------------
// post[t,d] = relu( sum_c xb[t,c]*wb[d,c] + b_enc[d] )
// Tile: 128 tokens x 128 dict per workgroup; 8 waves -> each 32x64 via
// 2x4 v_wmma_f32_16x16x32_bf16 accumulators; K-loop over 1024 in steps of 32.
// LDS rows padded to 40 ushorts (80B) -> conflict-free 16-lane frag reads.

#define LDSTRIDE 40

__global__ __launch_bounds__(256) void ae_encoder_wmma(
    const unsigned short* __restrict__ xb,   // 8192 x 1024 bf16
    const unsigned short* __restrict__ wb,   // 16384 x 1024 bf16
    const float* __restrict__ b_enc,
    float* __restrict__ post)                // 8192 x 16384 f32
{
  __shared__ __align__(16) unsigned short As[128 * LDSTRIDE];
  __shared__ __align__(16) unsigned short Bs[128 * LDSTRIDE];

  const int tid = threadIdx.x;
  const int n0 = blockIdx.x * 128;           // dict tile
  const int m0 = blockIdx.y * 128;           // token tile

  const int wid  = tid >> 5;
  const int lane = tid & 31;
  const int wm = (wid & 3) * 32;             // wave row offset in tile
  const int wn = (wid >> 2) * 64;            // wave col offset in tile
  const int lr = lane & 15;
  const int lh = lane >> 4;                  // lane half (0/1)

  const int ldr = tid >> 1;                  // 0..127 : row loaded by thread
  const int ldh = (tid & 1) * 16;            // 0 or 16: K-half loaded

  v8f acc[2][4] = {};

  for (int kt = 0; kt < AE_C / 32; ++kt) {
    const int c0 = kt * 32;
    // global -> LDS (each thread: 16 bf16 of A + 16 bf16 of B, 16B vectors)
    {
      const uint4* ga = (const uint4*)(xb + (size_t)(m0 + ldr) * AE_C + c0 + ldh);
      const uint4* gb = (const uint4*)(wb + (size_t)(n0 + ldr) * AE_C + c0 + ldh);
      uint4 a0 = ga[0], a1 = ga[1];
      uint4 b0 = gb[0], b1 = gb[1];
      uint4* la = (uint4*)&As[ldr * LDSTRIDE + ldh];
      la[0] = a0; la[1] = a1;
      uint4* lb = (uint4*)&Bs[ldr * LDSTRIDE + ldh];
      lb[0] = b0; lb[1] = b1;
    }
    __syncthreads();

    // A fragments: lane half 0 holds K {0..7,16..23}, half 1 {8..15,24..31}
    BFrag afr[2];
#pragma unroll
    for (int mt = 0; mt < 2; ++mt) {
      const unsigned short* ap = &As[(wm + mt * 16 + lr) * LDSTRIDE];
      afr[mt].u[0] = *(const uint4*)(ap + lh * 8);
      afr[mt].u[1] = *(const uint4*)(ap + 16 + lh * 8);
    }
    // B fragments: lane half 0 holds K {0..7,16..23}, half 1 {8..15,24..31} of col lr
#pragma unroll
    for (int nt = 0; nt < 4; ++nt) {
      BFrag bfr;
      const unsigned short* bp = &Bs[(wn + nt * 16 + lr) * LDSTRIDE];
      bfr.u[0] = *(const uint4*)(bp + lh * 8);
      bfr.u[1] = *(const uint4*)(bp + 16 + lh * 8);
#pragma unroll
      for (int mt = 0; mt < 2; ++mt) {
        acc[mt][nt] = __builtin_amdgcn_wmma_f32_16x16x32_bf16(
            false, afr[mt].v, false, bfr.v, (short)0, acc[mt][nt], false, false);
      }
    }
    __syncthreads();
  }

  // epilogue: + b_enc, ReLU, store (C/D layout: vgpr r -> M = r + lh*8, N = lr)
#pragma unroll
  for (int nt = 0; nt < 4; ++nt) {
    const int gcol = n0 + wn + nt * 16 + lr;
    const float bias = b_enc[gcol];
#pragma unroll
    for (int mt = 0; mt < 2; ++mt) {
#pragma unroll
      for (int r = 0; r < 8; ++r) {
        const int grow = m0 + wm + mt * 16 + lh * 8 + r;
        float v = acc[mt][nt][r] + bias;
        v = v > 0.0f ? v : 0.0f;
        post[(size_t)grow * AE_D + gcol] = v;
      }
    }
  }
}

// --------------------------- top-k + mask ----------------------------------
// One workgroup per window of 4 tokens. Window sums kept as bf16 in LDS
// (32KB) so static LDS stays < 64KB. 64 argmax iterations; then zero
// non-selected entries of the 4 rows in place.

__global__ __launch_bounds__(256) void ae_topk_mask(float* __restrict__ post,
                                                    int* __restrict__ idx_out) {
  __shared__ unsigned short sums[AE_D];        // 32 KB
  __shared__ unsigned mask[AE_D / 32];         // 2 KB
  __shared__ float rv[256];
  __shared__ int   ri[256];

  const int w   = blockIdx.x;
  const int tid = threadIdx.x;
  float* row0 = post + (size_t)w * AE_WIN * AE_D;

  for (int d = tid; d < AE_D; d += 256) {
    float s = row0[d] + row0[AE_D + d] + row0[2 * AE_D + d] + row0[3 * AE_D + d];
    sums[d] = f2bf(s);
  }
  for (int i = tid; i < AE_D / 32; i += 256) mask[i] = 0u;
  __syncthreads();

  const float NEG = -3.402823466e38f;
  for (int k = 0; k < AE_K; ++k) {
    float bv = NEG; int bi = 0;
    for (int d = tid; d < AE_D; d += 256) {
      float v = bf2f(sums[d]);
      if (v > bv) { bv = v; bi = d; }
    }
    rv[tid] = bv; ri[tid] = bi;
    __syncthreads();
    for (int s = 128; s > 0; s >>= 1) {
      if (tid < s && rv[tid + s] > rv[tid]) { rv[tid] = rv[tid + s]; ri[tid] = ri[tid + s]; }
      __syncthreads();
    }
    if (tid == 0) {
      int d = ri[0];
      idx_out[w * AE_K + k] = d;
      sums[d] = 0xFF80u;                       // bf16 -inf
      mask[d >> 5] |= (1u << (d & 31));
    }
    __syncthreads();
  }

  for (int j = 0; j < AE_WIN; ++j) {
    float* row = post + ((size_t)w * AE_WIN + j) * AE_D;
    for (int d = tid; d < AE_D; d += 256) {
      unsigned keep = (mask[d >> 5] >> (d & 31)) & 1u;
      float v = row[d];
      row[d] = keep ? v : 0.0f;
    }
  }
}

// --------------------------- sparse decoder --------------------------------
// rec[t,c] = b_dec[c] + sum_{k<64} enc[t, idx[w,k]] * Wt[idx[w,k], c]
// One workgroup per window; 256 threads over C (4 channels each), 4 tokens.

__global__ __launch_bounds__(256) void ae_decoder(const float* __restrict__ enc,
                                                  const int* __restrict__ idx,
                                                  const float* __restrict__ wt,
                                                  const float* __restrict__ b_dec,
                                                  float* __restrict__ rec) {
  __shared__ int   dsel[AE_K];
  __shared__ float act[AE_WIN][AE_K];

  const int w   = blockIdx.x;
  const int tid = threadIdx.x;

  if (tid < AE_K) dsel[tid] = idx[w * AE_K + tid];
  __syncthreads();
  {
    const int j = tid >> 6, k = tid & 63;      // 4 tokens x 64 features
    act[j][k] = enc[((size_t)w * AE_WIN + j) * AE_D + dsel[k]];
  }
  __syncthreads();

  float acc[AE_WIN][4];
#pragma unroll
  for (int cc = 0; cc < 4; ++cc) {
    const float b = b_dec[tid + cc * 256];
#pragma unroll
    for (int j = 0; j < AE_WIN; ++j) acc[j][cc] = b;
  }

  for (int k = 0; k < AE_K; ++k) {
    const float* row = wt + (size_t)dsel[k] * AE_C;
    const float w0 = row[tid], w1 = row[tid + 256], w2 = row[tid + 512], w3 = row[tid + 768];
#pragma unroll
    for (int j = 0; j < AE_WIN; ++j) {
      const float a = act[j][k];
      acc[j][0] += a * w0; acc[j][1] += a * w1; acc[j][2] += a * w2; acc[j][3] += a * w3;
    }
  }

#pragma unroll
  for (int j = 0; j < AE_WIN; ++j)
#pragma unroll
    for (int cc = 0; cc < 4; ++cc)
      rec[((size_t)w * AE_WIN + j) * AE_C + tid + cc * 256] = acc[j][cc];
}

// --------------------------- launch ----------------------------------------

extern "C" void kernel_launch(void* const* d_in, const int* in_sizes, int n_in,
                              void* d_out, int out_size, void* d_ws, size_t ws_size,
                              hipStream_t stream) {
  (void)in_sizes; (void)n_in; (void)out_size; (void)ws_size;
  const float* x     = (const float*)d_in[0];   // (16,512,1024)
  const float* W_enc = (const float*)d_in[1];   // (16384,1024)
  const float* b_enc = (const float*)d_in[2];   // (16384,)
  const float* W_dec = (const float*)d_in[3];   // (1024,16384)
  const float* b_dec = (const float*)d_in[4];   // (1024,)

  float* rec = (float*)d_out;                          // 8,388,608 f32
  float* enc = (float*)d_out + (size_t)AE_TOK * AE_C;  // 134,217,728 f32

  char* ws = (char*)d_ws;
  unsigned short* xb  = (unsigned short*)ws;                        // 16 MB
  unsigned short* wb  = (unsigned short*)(ws + ((size_t)16 << 20)); // 32 MB
  float*          wt  = (float*)(ws + ((size_t)48 << 20));          // 64 MB
  int*            idx = (int*)(ws + ((size_t)112 << 20));           // 512 KB

  ae_prep_x<<<dim3((AE_TOK * AE_C) / 256), dim3(256), 0, stream>>>(x, b_dec, xb, AE_TOK * AE_C);
  ae_prep_w<<<dim3((AE_D * AE_C) / 256), dim3(256), 0, stream>>>(W_enc, wb, AE_D * AE_C);
  ae_transpose_wdec<<<dim3(AE_D / 32, AE_C / 32), dim3(256), 0, stream>>>(W_dec, wt);
  ae_encoder_wmma<<<dim3(AE_D / 128, AE_TOK / 128), dim3(256), 0, stream>>>(xb, wb, b_enc, enc);
  ae_topk_mask<<<dim3(AE_NW), dim3(256), 0, stream>>>(enc, idx);
  ae_decoder<<<dim3(AE_NW), dim3(256), 0, stream>>>(enc, idx, wt, b_dec, rec);
}
